// FlowNetC_60954175865191
// MI455X (gfx1250) — matrix operand order repaired and checked by reading, not compiled, not run on hardware
//
#include <hip/hip_runtime.h>
#include <stdint.h>

#define IH 192
#define IW 192
#define HWPIX (IH*IW)
#define PH 194
#define PW 194
#define PP (PH*PW)          // padded pixel count (1-px zero border)
#define NB 4

typedef __attribute__((ext_vector_type(16))) __bf16 v16bf;
typedef __attribute__((ext_vector_type(8)))  float  v8f;

struct alignas(16) U4 { uint32_t x, y, z, w; };
union Frag { uint32_t u[8]; U4 q[2]; v16bf v; };

__device__ __forceinline__ float lrelu(float x) { return x >= 0.f ? x : 0.1f * x; }

// ---------------------------------------------------------------------------
// NCHW f32 -> padded NHWC bf16 (zero border). K contiguous for WMMA A-loads.
// ---------------------------------------------------------------------------
__global__ void cvt_nhwc_pad_bf16(const float* __restrict__ in, __bf16* __restrict__ out) {
    size_t idx = (size_t)blockIdx.x * blockDim.x + threadIdx.x;   // (b, c, ppad)
    size_t total = (size_t)NB * 64 * PP;
    if (idx >= total) return;
    int pp = (int)(idx % PP);
    int c  = (int)((idx / PP) % 64);
    int b  = (int)(idx / ((size_t)PP * 64));
    int py = pp / PW, px = pp % PW;
    float v = 0.f;
    if (py >= 1 && py <= IH && px >= 1 && px <= IW)
        v = in[(size_t)(b * 64 + c) * HWPIX + (py - 1) * IW + (px - 1)];
    out[((size_t)(b * PP + pp)) * 64 + c] = (__bf16)v;
}

// ---------------------------------------------------------------------------
// Repack OIHW f32 weights into per-lane WMMA B-fragment stream (bf16).
// Region = (tap, kchunk, ocTile): 32 lanes x 8 VGPRs x 2 bf16 = 1024 B.
// B layout (32x16 bf16): lane -> n=lane&15, k=(v>>2)*16 + (lane>>4)*8 + 2(v&3) (+1 in hi16)
// ---------------------------------------------------------------------------
__global__ void repack_weights(const float* __restrict__ w, __bf16* __restrict__ dst,
                               int Cin, int Cout) {
    int chunks = Cin / 32, nOcT = Cout / 16;
    int total = 9 * chunks * nOcT * 256;               // 32-bit pairs
    int idx = blockIdx.x * blockDim.x + threadIdx.x;
    if (idx >= total) return;
    int v    = idx & 7;
    int lane = (idx >> 3) & 31;
    int r    = idx >> 8;
    int ocT   = r % nOcT;
    int chunk = (r / nOcT) % chunks;
    int tap   = r / (nOcT * chunks);
    int k  = (v >> 2) * 16 + (lane >> 4) * 8 + 2 * (v & 3);
    int ic = chunk * 32 + k;
    int oc = ocT * 16 + (lane & 15);
    int ky = tap / 3, kx = tap % 3;
    float w0 = w[((oc * Cin + ic) * 3 + ky) * 3 + kx];
    float w1 = w[((oc * Cin + ic + 1) * 3 + ky) * 3 + kx];
    dst[idx * 2 + 0] = (__bf16)w0;   // bits[15:0]  = K
    dst[idx * 2 + 1] = (__bf16)w1;   // bits[31:16] = K+1
}

// ---------------------------------------------------------------------------
// 3x3 conv, implicit GEMM on v_wmma_f32_16x16x32_bf16, + bias + leaky ReLU.
// Input is padded NHWC bf16 -> every tap in-bounds -> unrolled K-loop is
// pure constant-offset global_load_b128 clauses + WMMA (no exec masking).
// M = 16 pixels along x, N = 16 oc, K = 9 taps x CIN. 8 waves / block.
// ---------------------------------------------------------------------------
template <int CIN, int COUT, bool OUT_PADDED_BF16>
__global__ void conv3x3_wmma(const __bf16* __restrict__ in,   // padded NHWC bf16
                             const __bf16* __restrict__ wpk,  // repacked B fragments
                             const float*  __restrict__ bias, // [COUT]
                             void* __restrict__ outv) {       // padded bf16 or plain f32
    constexpr int chunks = CIN / 32;
    constexpr int nOcT   = COUT / 16;
    constexpr int XS     = 8 / nOcT;

    int wave = threadIdx.x >> 5;
    int lane = threadIdx.x & 31;
    int ocT  = wave % nOcT;
    int xsub = wave / nOcT;
    int x0 = (blockIdx.x * XS + xsub) * 16;
    int y  = blockIdx.y;
    int b  = blockIdx.z;

    int m  = lane & 15;     // A row (pixel in tile)
    int kh = lane >> 4;     // K-half select

    // Per-thread bases; everything else is compile-time immediate offsets.
    const __bf16* pbase = in +
        ((size_t)(b * PP + (y + 1) * PW + (x0 + m + 1))) * CIN + kh * 8;
    const __bf16* wbase = wpk + (size_t)ocT * 512 + lane * 16;

    v8f acc = {0.f, 0.f, 0.f, 0.f, 0.f, 0.f, 0.f, 0.f};

    #pragma unroll
    for (int tap = 0; tap < 9; ++tap) {
        const int dy = tap / 3 - 1, dx = tap % 3 - 1;
        const int poff = (dy * PW + dx) * CIN;               // constant
        #pragma unroll
        for (int chunk = 0; chunk < chunks; ++chunk) {
            Frag a;
            a.q[0] = *(const U4*)(pbase + poff + chunk * 32);        // K 0..7 (half)
            a.q[1] = *(const U4*)(pbase + poff + chunk * 32 + 16);   // K 16..23 (half)
            Frag bf;
            const __bf16* bp = wbase + (size_t)(tap * chunks + chunk) * nOcT * 512;
            bf.q[0] = *(const U4*)(bp);
            bf.q[1] = *(const U4*)(bp + 8);
            acc = __builtin_amdgcn_wmma_f32_16x16x32_bf16(
                false, a.v, false, bf.v, (short)0, acc, false, false);
        }
    }

    int oc = ocT * 16 + (lane & 15);            // D: lane&15 = column n = oc
    float bv = bias[oc];
    #pragma unroll
    for (int v = 0; v < 8; ++v) {               // D: row m = v + 8*(lane>>4)
        int row = v + 8 * kh;
        float val = lrelu(acc[v] + bv);
        if (OUT_PADDED_BF16) {
            size_t o = ((size_t)(b * PP + (y + 1) * PW + (x0 + row + 1))) * COUT + oc;
            ((__bf16*)outv)[o] = (__bf16)val;
        } else {
            size_t o = ((size_t)(b * HWPIX + y * IW + (x0 + row))) * COUT + oc;
            ((float*)outv)[o] = val;
        }
    }
}

// ---------------------------------------------------------------------------
// Fused: redir(1x1,128->32,lrelu) + correlation(441 disp, /128, lrelu)
//        + final 3x3 conv 473->2 (zero pad), never materializing corr.
// Block = 256 threads = 16x16 halo pixels; 14x14 output region per block.
// ---------------------------------------------------------------------------
#define X1_STRIDE 132
__global__ void corr_pred_kernel(const float* __restrict__ c2a,   // NHWC f32 [B,H,W,128]
                                 const float* __restrict__ c2b,
                                 const float* __restrict__ redir_w,  // [32][128]
                                 const float* __restrict__ redir_b,  // [32]
                                 const float* __restrict__ pred_w,   // [2][473][3][3]
                                 float* __restrict__ out) {          // [B,2,H,W]
    extern __shared__ float smem[];
    float* sx1 = smem;                     // 256 * 132 floats
    float* sc  = smem + 256 * X1_STRIDE;   // 256 floats

    int t  = threadIdx.x;
    int hx = t & 15, hy = t >> 4;
    int ox = blockIdx.x * 14, oy = blockIdx.y * 14;
    int b  = blockIdx.z;
    int gx = ox + hx - 1, gy = oy + hy - 1;
    bool inimg = (gx >= 0) & (gx < IW) & (gy >= 0) & (gy < IH);

    if (inimg) {
        const float* p = c2a + ((size_t)(b * HWPIX + gy * IW + gx)) * 128;
        #pragma unroll
        for (int i = 0; i < 32; ++i)
            *(float4*)&sx1[t * X1_STRIDE + 4 * i] = *(const float4*)&p[4 * i];
    } else {
        #pragma unroll
        for (int i = 0; i < 32; ++i)
            *(float4*)&sx1[t * X1_STRIDE + 4 * i] = float4{0.f, 0.f, 0.f, 0.f};
    }
    __syncthreads();

    bool inner = (hx >= 1) & (hx <= 14) & (hy >= 1) & (hy <= 14) & (gx < IW) & (gy < IH);
    float acc0 = 0.f, acc1 = 0.f;

    for (int ch = 0; ch < 473; ++ch) {
        float val = 0.f;
        if (ch < 32) {                                     // redir channel
            if (inimg) {
                const float* w = redir_w + ch * 128;
                float s = redir_b[ch];
                #pragma unroll 8
                for (int i = 0; i < 32; ++i) {
                    float4 a  = *(const float4*)&sx1[t * X1_STRIDE + 4 * i];
                    float4 ww = *(const float4*)&w[4 * i];
                    s += a.x * ww.x + a.y * ww.y + a.z * ww.z + a.w * ww.w;
                }
                val = lrelu(s);
            }
        } else {                                           // correlation channel
            int d  = ch - 32;
            int sy = gy + 2 * (d / 21) - 20;
            int sx = gx + 2 * (d % 21) - 20;
            if (inimg & (sy >= 0) & (sy < IH) & (sx >= 0) & (sx < IW)) {
                const float* p = c2b + ((size_t)(b * HWPIX + sy * IW + sx)) * 128;
                float s = 0.f;
                #pragma unroll 8
                for (int i = 0; i < 32; ++i) {
                    float4 a = *(const float4*)&sx1[t * X1_STRIDE + 4 * i];
                    float4 x = *(const float4*)&p[4 * i];
                    s += a.x * x.x + a.y * x.y + a.z * x.z + a.w * x.w;
                }
                val = lrelu(s * (1.f / 128.f));
            }
        }
        sc[t] = val;
        __syncthreads();
        if (inner) {
            #pragma unroll
            for (int ky = 0; ky < 3; ++ky)
                #pragma unroll
                for (int kx = 0; kx < 3; ++kx) {
                    float v = sc[(hy - 1 + ky) * 16 + (hx - 1 + kx)];
                    acc0 += v * pred_w[((0 * 473 + ch) * 3 + ky) * 3 + kx];
                    acc1 += v * pred_w[((1 * 473 + ch) * 3 + ky) * 3 + kx];
                }
        }
        __syncthreads();
    }

    if (inner) {
        out[(size_t)(b * 2 + 0) * HWPIX + gy * IW + gx] = acc0;
        out[(size_t)(b * 2 + 1) * HWPIX + gy * IW + gx] = acc1;
    }
}

// ---------------------------------------------------------------------------
extern "C" void kernel_launch(void* const* d_in, const int* in_sizes, int n_in,
                              void* d_out, int out_size, void* d_ws, size_t ws_size,
                              hipStream_t stream) {
    (void)in_sizes; (void)n_in; (void)out_size; (void)ws_size;
    const float* pred = (const float*)d_in[0];
    const float* ref  = (const float*)d_in[1];
    const float* c1w  = (const float*)d_in[2];
    const float* c1b  = (const float*)d_in[3];
    const float* c2w  = (const float*)d_in[4];
    const float* c2bi = (const float*)d_in[5];
    const float* rw   = (const float*)d_in[6];
    const float* rb   = (const float*)d_in[7];
    const float* pw   = (const float*)d_in[8];
    float* out = (float*)d_out;

    char* ws = (char*)d_ws;
    size_t off = 0;
    auto alloc = [&](size_t bytes) { size_t p = off; off = (off + bytes + 255) & ~(size_t)255; return p; };
    const size_t tpadBytes = (size_t)NB * PP * 64 * sizeof(__bf16);          // ~19.3 MB
    __bf16* w1pk = (__bf16*)(ws + alloc((size_t)9 * 2 * 4 * 1024));          // 72 KB
    __bf16* w2pk = (__bf16*)(ws + alloc((size_t)9 * 2 * 8 * 1024));          // 144 KB
    __bf16* t0   = (__bf16*)(ws + alloc(tpadBytes));                         // padded (reused)
    __bf16* t1   = (__bf16*)(ws + alloc(tpadBytes));                         // padded (reused)
    float*  c2a  = (float*)(ws + alloc((size_t)NB * HWPIX * 128 * 4));       // 75.5 MB
    float*  c2b  = (float*)(ws + alloc((size_t)NB * HWPIX * 128 * 4));       // 75.5 MB

    // Weight repack (bf16 B-fragment streams)
    repack_weights<<<(9 * 2 * 4 * 256 + 255) / 256, 256, 0, stream>>>(c1w, w1pk, 64, 64);
    repack_weights<<<(9 * 2 * 8 * 256 + 255) / 256, 256, 0, stream>>>(c2w, w2pk, 64, 128);

    // Zero t1 once: conv1 writes only the interior; border must be zero.
    // (Both branches rewrite the interior; border stays zero.)
    (void)hipMemsetAsync(t1, 0, tpadBytes, stream);

    size_t cvtN = (size_t)NB * 64 * PP;
    dim3 cvtG((unsigned)((cvtN + 255) / 256));

    // pred branch: t0 -> conv1 -> t1 -> conv2 -> c2a
    cvt_nhwc_pad_bf16<<<cvtG, 256, 0, stream>>>(pred, t0);
    conv3x3_wmma<64, 64, true ><<<dim3(6, IH, NB), 256, 0, stream>>>(t0, w1pk, c1b, t1);
    conv3x3_wmma<64, 128, false><<<dim3(12, IH, NB), 256, 0, stream>>>(t1, w2pk, c2bi, c2a);

    // ref branch (reuses t0/t1): -> c2b
    cvt_nhwc_pad_bf16<<<cvtG, 256, 0, stream>>>(ref, t0);
    conv3x3_wmma<64, 64, true ><<<dim3(6, IH, NB), 256, 0, stream>>>(t0, w1pk, c1b, t1);
    conv3x3_wmma<64, 128, false><<<dim3(12, IH, NB), 256, 0, stream>>>(t1, w2pk, c2bi, c2b);

    // Fused redir + correlation + final 3x3 prediction conv
    int shmem = (256 * X1_STRIDE + 256) * (int)sizeof(float);   // 136192 B < 320 KB/WGP
    (void)hipFuncSetAttribute(reinterpret_cast<const void*>(corr_pred_kernel),
                              hipFuncAttributeMaxDynamicSharedMemorySize, shmem);
    corr_pred_kernel<<<dim3(14, 14, NB), 256, shmem, stream>>>(c2a, c2b, rw, rb, pw, out);
}
